// GraphNetwork_31877247271096
// MI455X (gfx1250) — matrix operand order repaired and checked by reading, not compiled, not run on hardware
//
#include <hip/hip_runtime.h>
#include <hip/hip_bf16.h>

typedef __attribute__((ext_vector_type(16))) _Float16 v16h;
typedef __attribute__((ext_vector_type(8)))  _Float16 v8h;
typedef __attribute__((ext_vector_type(8)))  float    v8f;
typedef __attribute__((ext_vector_type(4)))  float    f32x4;

#define BATCH   65536
#define FDIM    256
#define HID1    512
#define HID2    512
#define LAT     256
#define TB      16
#define NWG     (BATCH / TB)          // 4096 workgroups
#define LPITCH  520                   // halves per LDS row (512 + 8 pad)

// fragment-ordered f16 weights in ws
#define W1F_ELEMS (FDIM * HID1)                       // 131072
#define W2F_ELEMS (HID1 * HID2)                       // 262144
#define W3F_ELEMS (HID2 * LAT)                        // 131072
#define WF_TOTAL  (W1F_ELEMS + W2F_ELEMS + W3F_ELEMS) // 524288 halves
#define PSUM_OFF  ((size_t)WF_TOTAL * 2)              // bytes
#define PSQ_OFF   (PSUM_OFF + (size_t)NWG * LAT * 4)
#define MU_OFF    (PSQ_OFF  + (size_t)NWG * LAT * 4)
#define RS_OFF    (MU_OFF   + (size_t)LAT * 4)

// ---------------------------------------------------------------------------
// Rearrange fp32 weights into f16 WMMA B-fragment order:
// tile t = kw*ntiles + nt; within tile, lane L stores 16 halves where
// element h maps to W[kw*32 + (h%8) + 8*(L/16) + 16*(h/8)][nt*16 + L%16].
// ---------------------------------------------------------------------------
__global__ __launch_bounds__(256) void prep_weights(
    const float* __restrict__ W1, const float* __restrict__ W2,
    const float* __restrict__ W3, _Float16* __restrict__ wf)
{
    int t = blockIdx.x * 256 + threadIdx.x;
    const float* src; _Float16* dst; int ntiles, ncols, e;
    if (t < W1F_ELEMS)                  { src = W1; dst = wf;                         ntiles = 32; ncols = HID1; e = t; }
    else if (t < W1F_ELEMS + W2F_ELEMS) { src = W2; dst = wf + W1F_ELEMS;            ntiles = 32; ncols = HID2; e = t - W1F_ELEMS; }
    else                                { src = W3; dst = wf + W1F_ELEMS + W2F_ELEMS; ntiles = 16; ncols = LAT;  e = t - W1F_ELEMS - W2F_ELEMS; }
    int tile = e >> 9;
    int lane = (e >> 4) & 31;
    int h    = e & 15;
    int kw   = tile / ntiles;
    int nt   = tile % ntiles;
    int K    = kw * 32 + (h & 7) + 8 * (lane >> 4) + 16 * (h >> 3);
    int N    = nt * 16 + (lane & 15);
    dst[e] = (_Float16)src[(size_t)K * ncols + N];
}

// ---------------------------------------------------------------------------
// Fused: edge-sum -> MLP(W1,relu) -> MLP(W2,relu) -> edge-aggregate -> W3
// One workgroup = 16 batch rows. 8 waves, all GEMMs via v_wmma_f32_16x16x32_f16.
// ---------------------------------------------------------------------------
__global__ __launch_bounds__(256) void fused_gn(
    const float* __restrict__ x,
    const float* __restrict__ b1, const float* __restrict__ b2,
    const float* __restrict__ b3,
    const _Float16* __restrict__ wf,
    float* __restrict__ psum, float* __restrict__ psq,
    float* __restrict__ out)
{
    __shared__ __align__(16) _Float16 lds[48 * LPITCH];   // h1 (48x512), reused for agg (16x512)

    const int wave = threadIdx.x >> 5;
    const int lane = threadIdx.x & 31;
    const int lr   = lane & 15;          // row / col within 16
    const int lh   = lane >> 4;          // half-wave select
    const int b0   = blockIdx.x * TB;

    const _Float16* W1f = wf;
    const _Float16* W2f = wf + W1F_ELEMS;
    const _Float16* W3f = wf + W1F_ELEMS + W2F_ELEMS;

    union AFrag { v16h v; v8h p[2]; _Float16 h[16]; };

    const int pi[3] = {0, 0, 1};
    const int pj[3] = {1, 2, 2};

    // ---------------- Stage 1: h1 = relu(E @ W1 + b1) -> LDS ----------------
    for (int rt = 0; rt < 3; ++rt) {
        const float* xi = x + ((size_t)(b0 + lr) * 3 + pi[rt]) * FDIM;
        const float* xj = x + ((size_t)(b0 + lr) * 3 + pj[rt]) * FDIM;
        v8f c[4];
        #pragma unroll
        for (int q = 0; q < 4; ++q) {
            float bv = b1[(wave * 4 + q) * 16 + lr];
            #pragma unroll
            for (int r = 0; r < 8; ++r) c[q][r] = bv;
        }
        #pragma unroll
        for (int kw = 0; kw < 8; ++kw) {
            const int cb = kw * 32 + 8 * lh;
            f32x4 xa0 = *(const f32x4*)(xi + cb);
            f32x4 xa1 = *(const f32x4*)(xi + cb + 4);
            f32x4 xa2 = *(const f32x4*)(xi + cb + 16);
            f32x4 xa3 = *(const f32x4*)(xi + cb + 20);
            f32x4 xb0 = *(const f32x4*)(xj + cb);
            f32x4 xb1 = *(const f32x4*)(xj + cb + 4);
            f32x4 xb2 = *(const f32x4*)(xj + cb + 16);
            f32x4 xb3 = *(const f32x4*)(xj + cb + 20);
            AFrag a;
            #pragma unroll
            for (int t = 0; t < 4; ++t) {
                a.h[t]      = (_Float16)(xa0[t] + xb0[t]);
                a.h[4 + t]  = (_Float16)(xa1[t] + xb1[t]);
                a.h[8 + t]  = (_Float16)(xa2[t] + xb2[t]);
                a.h[12 + t] = (_Float16)(xa3[t] + xb3[t]);
            }
            #pragma unroll
            for (int q = 0; q < 4; ++q) {
                const int nt = wave * 4 + q;
                v16h bf = *(const v16h*)(W1f + ((size_t)(kw * 32 + nt) * 512 + lane * 16));
                c[q] = __builtin_amdgcn_wmma_f32_16x16x32_f16(
                           false, a.v, false, bf, (short)0, c[q], false, false);
            }
        }
        #pragma unroll
        for (int q = 0; q < 4; ++q) {
            const int col = (wave * 4 + q) * 16 + lr;
            #pragma unroll
            for (int r = 0; r < 8; ++r) {
                float v = c[q][r];
                lds[(rt * 16 + r + 8 * lh) * LPITCH + col] = (_Float16)(v > 0.f ? v : 0.f);
            }
        }
    }
    __syncthreads();

    // ------- Stage 2: agg = sum_e relu(h1_e @ W2 + b2), kept in regs -------
    float agg[4][8];
    #pragma unroll
    for (int q = 0; q < 4; ++q)
        #pragma unroll
        for (int r = 0; r < 8; ++r) agg[q][r] = 0.f;

    for (int rt = 0; rt < 3; ++rt) {
        v8f c[4];
        #pragma unroll
        for (int q = 0; q < 4; ++q) {
            float bv = b2[(wave * 4 + q) * 16 + lr];
            #pragma unroll
            for (int r = 0; r < 8; ++r) c[q][r] = bv;
        }
        #pragma unroll
        for (int kw = 0; kw < 16; ++kw) {
            const int cb = kw * 32 + 8 * lh;
            const _Float16* rowp = &lds[(rt * 16 + lr) * LPITCH + cb];
            AFrag a;
            a.p[0] = *(const v8h*)(rowp);
            a.p[1] = *(const v8h*)(rowp + 16);
            #pragma unroll
            for (int q = 0; q < 4; ++q) {
                const int nt = wave * 4 + q;
                v16h bf = *(const v16h*)(W2f + ((size_t)(kw * 32 + nt) * 512 + lane * 16));
                c[q] = __builtin_amdgcn_wmma_f32_16x16x32_f16(
                           false, a.v, false, bf, (short)0, c[q], false, false);
            }
        }
        #pragma unroll
        for (int q = 0; q < 4; ++q)
            #pragma unroll
            for (int r = 0; r < 8; ++r) {
                float v = c[q][r];
                agg[q][r] += (v > 0.f ? v : 0.f);
            }
    }
    __syncthreads();                 // all stage-2 reads of h1 done
    #pragma unroll
    for (int q = 0; q < 4; ++q) {
        const int col = (wave * 4 + q) * 16 + lr;
        #pragma unroll
        for (int r = 0; r < 8; ++r)
            lds[(r + 8 * lh) * LPITCH + col] = (_Float16)agg[q][r];
    }
    __syncthreads();

    // ----------- Stage 3: z = agg @ W3 + b3 ; write z + BN partials --------
    v8f cz[2];
    #pragma unroll
    for (int q = 0; q < 2; ++q) {
        float bv = b3[(wave * 2 + q) * 16 + lr];
        #pragma unroll
        for (int r = 0; r < 8; ++r) cz[q][r] = bv;
    }
    #pragma unroll
    for (int kw = 0; kw < 16; ++kw) {
        const int cb = kw * 32 + 8 * lh;
        const _Float16* rowp = &lds[lr * LPITCH + cb];
        AFrag a;
        a.p[0] = *(const v8h*)(rowp);
        a.p[1] = *(const v8h*)(rowp + 16);
        #pragma unroll
        for (int q = 0; q < 2; ++q) {
            const int nt = wave * 2 + q;
            v16h bf = *(const v16h*)(W3f + ((size_t)(kw * 16 + nt) * 512 + lane * 16));
            cz[q] = __builtin_amdgcn_wmma_f32_16x16x32_f16(
                        false, a.v, false, bf, (short)0, cz[q], false, false);
        }
    }
    #pragma unroll
    for (int q = 0; q < 2; ++q) {
        const int col = (wave * 2 + q) * 16 + lr;
        float s = 0.f, ss = 0.f;
        #pragma unroll
        for (int r = 0; r < 8; ++r) {
            float v = cz[q][r];
            out[(size_t)(b0 + r + 8 * lh) * LAT + col] = v;
            s += v; ss += v * v;
        }
        s  += __shfl_xor(s, 16);
        ss += __shfl_xor(ss, 16);
        if (lh == 0) {
            psum[(size_t)blockIdx.x * LAT + col] = s;
            psq [(size_t)blockIdx.x * LAT + col] = ss;
        }
    }
}

// ---------------------------------------------------------------------------
// Per-feature reduction of partials -> mu, rsig (one block per feature)
// ---------------------------------------------------------------------------
__global__ __launch_bounds__(256) void stats_kernel(
    const float* __restrict__ psum, const float* __restrict__ psq,
    float* __restrict__ mu, float* __restrict__ rsig)
{
    const int col = blockIdx.x;
    const int t   = threadIdx.x;
    float s = 0.f, q = 0.f;
    for (int wg = t; wg < NWG; wg += 256) {
        s += psum[(size_t)wg * LAT + col];
        q += psq [(size_t)wg * LAT + col];
    }
    __shared__ float sh[256], shq[256];
    sh[t] = s; shq[t] = q;
    __syncthreads();
    for (int off = 128; off > 0; off >>= 1) {
        if (t < off) { sh[t] += sh[t + off]; shq[t] += shq[t + off]; }
        __syncthreads();
    }
    if (t == 0) {
        float m = sh[0] * (1.f / (float)BATCH);
        float v = shq[0] * (1.f / (float)BATCH) - m * m;
        mu[col]   = m;
        rsig[col] = rsqrtf(v + 1e-5f);
    }
}

// ---------------------------------------------------------------------------
// In-place batchnorm apply on d_out
// ---------------------------------------------------------------------------
__global__ __launch_bounds__(256) void norm_kernel(
    float* __restrict__ out, const float* __restrict__ mu,
    const float* __restrict__ rsig, const float* __restrict__ gamma,
    const float* __restrict__ beta)
{
    size_t idx = (size_t)blockIdx.x * 256 + threadIdx.x;
    int col = (int)(idx & (LAT - 1));
    out[idx] = (out[idx] - mu[col]) * rsig[col] * gamma[col] + beta[col];
}

// ---------------------------------------------------------------------------
extern "C" void kernel_launch(void* const* d_in, const int* in_sizes, int n_in,
                              void* d_out, int out_size, void* d_ws, size_t ws_size,
                              hipStream_t stream)
{
    const float* x     = (const float*)d_in[0];
    const float* W1    = (const float*)d_in[1];
    const float* b1    = (const float*)d_in[2];
    const float* W2    = (const float*)d_in[3];
    const float* b2    = (const float*)d_in[4];
    const float* W3    = (const float*)d_in[5];
    const float* b3    = (const float*)d_in[6];
    const float* gamma = (const float*)d_in[7];
    const float* beta  = (const float*)d_in[8];

    char* ws = (char*)d_ws;
    _Float16* wf  = (_Float16*)ws;
    float* psum   = (float*)(ws + PSUM_OFF);
    float* psq    = (float*)(ws + PSQ_OFF);
    float* mu     = (float*)(ws + MU_OFF);
    float* rsig   = (float*)(ws + RS_OFF);
    float* out    = (float*)d_out;

    prep_weights<<<WF_TOTAL / 256, 256, 0, stream>>>(W1, W2, W3, wf);
    fused_gn<<<NWG, 256, 0, stream>>>(x, b1, b2, b3, wf, psum, psq, out);
    stats_kernel<<<LAT, 256, 0, stream>>>(psum, psq, mu, rsig);
    norm_kernel<<<(BATCH * LAT) / 256, 256, 0, stream>>>(out, mu, rsig, gamma, beta);
}
